// MultiEdit_12876311954011
// MI455X (gfx1250) — matrix operand order repaired and checked by reading, not compiled, not run on hardware
//
#include <hip/hip_runtime.h>

#define HSIZE 256
#define MLPSZ 512

// ---------------------------------------------------------------------------
// WMMA types / helpers (CDNA5 gfx1250, wave32)
// ---------------------------------------------------------------------------
typedef __attribute__((ext_vector_type(16))) __bf16 v16bf;
typedef __attribute__((ext_vector_type(8)))  float  v8f;

union ABReg { unsigned int u[8]; v16bf v; };

__device__ __forceinline__ unsigned short f32_to_bf16(float f) {
    unsigned int u = __float_as_uint(f);
    u += 0x7FFFu + ((u >> 16) & 1u);          // round-to-nearest-even
    return (unsigned short)(u >> 16);
}
__device__ __forceinline__ unsigned int pack2_bf16(float lo, float hi) {
    return (unsigned int)f32_to_bf16(lo) | ((unsigned int)f32_to_bf16(hi) << 16);
}

// A-tile (16x32 bf16, MxK) from row-major bf16 buffer (ldk elems/row).
// ISA layout: lane r=l&15 -> row M=r; half h=l>>4; VGPR j<4 holds K=8h+2j(,+1),
// VGPR j>=4 holds K=16+8h+2(j-4)(,+1)  => pair indices {4h..4h+3, 8+4h..8+4h+3}.
__device__ __forceinline__ v16bf load_a_tile(const unsigned short* A, int ldk,
                                             int kbase, int lane) {
    const int r = lane & 15, h = lane >> 4;
    const unsigned int* p = (const unsigned int*)(A + (size_t)r * ldk + kbase);
    ABReg a;
#pragma unroll
    for (int j = 0; j < 4; ++j) { a.u[j] = p[4*h + j]; a.u[4+j] = p[8 + 4*h + j]; }
    return a.v;
}

// B-tile (32x16 bf16, KxN) from K-contiguous (transposed) weights Wt[Npad][K].
// ISA layout: lane c=l&15 -> col N=c; half h: VGPR j<4 holds K=8h+2j(,+1),
// VGPR j>=4 holds K=16+8h+2(j-4)(,+1) (same k order as A).
__device__ __forceinline__ v16bf load_b_tile(const unsigned short* Wt, int K,
                                             int kbase, int nbase, int lane) {
    const int c = lane & 15, h = lane >> 4;
    const unsigned int* p = (const unsigned int*)(Wt + (size_t)(nbase + c) * K + kbase);
    ABReg b;
#pragma unroll
    for (int j = 0; j < 4; ++j) { b.u[j] = p[4*h + j]; b.u[4+j] = p[8 + 4*h + j]; }
    return b.v;
}

__device__ __forceinline__ v8f wmma_bf16(v16bf a, v16bf b, v8f c) {
    return __builtin_amdgcn_wmma_f32_16x16x32_bf16(false, a, false, b,
                                                   (short)0, c, false, false);
}

// ---------------------------------------------------------------------------
// Weight prep: fp32 [K][N] -> bf16 transposed+padded [Npad][K]
// ---------------------------------------------------------------------------
__global__ void transpose_pad_bf16(const float* __restrict__ src,
                                   unsigned short* __restrict__ dst,
                                   int K, int Nsrc, int Npad) {
    int i = blockIdx.x * blockDim.x + threadIdx.x;
    if (i >= K * Npad) return;
    int n = i / K, k = i - n * K;
    float v = (n < Nsrc) ? src[(size_t)k * Nsrc + n] : 0.f;
    dst[i] = f32_to_bf16(v);
}

// Combined [W_vv; W_vc] (K=512) transposed to bf16 [256][512]
__global__ void build_wcomb_bf16(const float* __restrict__ Wvv,
                                 const float* __restrict__ Wvc,
                                 unsigned short* __restrict__ dst) {
    int i = blockIdx.x * blockDim.x + threadIdx.x;
    if (i >= 256 * 512) return;
    int n = i >> 9, k = i & 511;
    float v = (k < 256) ? Wvv[k * 256 + n] : Wvc[(k - 256) * 256 + n];
    dst[i] = f32_to_bf16(v);
}

__global__ void zero_f32(float* p, int n) {
    int i = blockIdx.x * blockDim.x + threadIdx.x;
    if (i < n) p[i] = 0.f;
}

// ---------------------------------------------------------------------------
// Stage 1: ha = relu([ha_prev|c_atom] @ Wcomb); ha[0]=0; fused segment_sum->hm
// Block = 64 atom rows staged in LDS (bf16); each wave owns 4 col-tiles and
// reuses every B tile across 4 row tiles. B is double-buffered; A re-read
// from LDS each step (keeps VGPR count low -> multi-wave occupancy).
// ---------------------------------------------------------------------------
__global__ __launch_bounds__(128) void ha_kernel(
    const float* __restrict__ ha_prev, const float* __restrict__ c_atom,
    const unsigned short* __restrict__ Wcomb,   // [256][512] bf16
    const int* __restrict__ mol_id,
    unsigned short* __restrict__ ha_bf, float* __restrict__ hm,
    int n_atoms)
{
    __shared__ __align__(16) unsigned short sA[64 * 512];   // 64 KB
    const int row0 = blockIdx.x * 64;
    const int t = threadIdx.x;
    // stage+convert: 4096 chunks of 8 fp32 -> 8 bf16 (uint4)
    for (int c = t; c < 4096; c += 128) {
        int r = c >> 6;
        int e = (c & 63) << 3;
        int grow = row0 + r;
        int gro  = grow < n_atoms ? grow : n_atoms - 1;
        const float* src = (e < HSIZE)
            ? (ha_prev + (size_t)gro * HSIZE + e)
            : (c_atom  + (size_t)gro * HSIZE + (e - HSIZE));
        float4 f0 = ((const float4*)src)[0];
        float4 f1 = ((const float4*)src)[1];
        if (grow >= n_atoms) { f0 = make_float4(0,0,0,0); f1 = f0; }
        uint4 pk;
        pk.x = pack2_bf16(f0.x, f0.y); pk.y = pack2_bf16(f0.z, f0.w);
        pk.z = pack2_bf16(f1.x, f1.y); pk.w = pack2_bf16(f1.z, f1.w);
        *((uint4*)(sA + r * 512 + e)) = pk;
    }
    __syncthreads();
    const int wave = t >> 5, lane = t & 31;
    const int n_lo = lane & 15, h = lane >> 4;
    for (int cti = 0; cti < 4; ++cti) {
        const int ct = wave * 4 + cti;
        v8f acc[4];
#pragma unroll
        for (int m = 0; m < 4; ++m) acc[m] = (v8f){};
        v16bf b_cur = load_b_tile(Wcomb, 512, 0, ct * 16, lane);
#pragma unroll 1
        for (int ks = 0; ks < 512 - 32; ks += 32) {
            v16bf b_next = load_b_tile(Wcomb, 512, ks + 32, ct * 16, lane);
#pragma unroll
            for (int m = 0; m < 4; ++m) {
                v16bf a = load_a_tile(sA + m * 16 * 512, 512, ks, lane);
                acc[m] = wmma_bf16(a, b_cur, acc[m]);
            }
            b_cur = b_next;
        }
#pragma unroll
        for (int m = 0; m < 4; ++m) {           // peeled last k-step
            v16bf a = load_a_tile(sA + m * 16 * 512, 512, 512 - 32, lane);
            acc[m] = wmma_bf16(a, b_cur, acc[m]);
        }
        const int n = ct * 16 + n_lo;
#pragma unroll
        for (int m = 0; m < 4; ++m) {
#pragma unroll
            for (int v = 0; v < 8; ++v) {
                int grow = row0 + m * 16 + v + 8 * h;
                if (grow < n_atoms) {
                    float val = acc[m][v];
                    val = val > 0.f ? val : 0.f;
                    if (grow == 0) val = 0.f;       // padding atom
                    ha_bf[(size_t)grow * HSIZE + n] = f32_to_bf16(val);
                    atomicAdd(&hm[(size_t)mol_id[grow] * HSIZE + n], val);
                }
            }
        }
    }
}

// ---------------------------------------------------------------------------
// Fused MLP machinery on a 64-row block:
//   layer1: each wave owns 8 hidden col-tiles; B double-buffered, reused
//           across 4 row tiles; A streamed from LDS
//   layer2: each wave owns its 16-row tile against the padded [16][512] W2
// ---------------------------------------------------------------------------
template<int K1>
__device__ __forceinline__ void mlp_layer1_block(
    const unsigned short* sA,        // [64][K1] bf16 staged rows
    unsigned short* sH,              // [64][512] bf16 hidden
    const unsigned short* W1t,       // [512][K1]
    const float* b1, int wave, int lane)
{
    const int n_lo = lane & 15, h = lane >> 4;
    for (int cti = 0; cti < 8; ++cti) {
        const int ct = wave * 8 + cti;
        v8f acc[4];
#pragma unroll
        for (int m = 0; m < 4; ++m) acc[m] = (v8f){};
        v16bf b_cur = load_b_tile(W1t, K1, 0, ct * 16, lane);
#pragma unroll 1
        for (int ks = 0; ks < K1 - 32; ks += 32) {
            v16bf b_next = load_b_tile(W1t, K1, ks + 32, ct * 16, lane);
#pragma unroll
            for (int m = 0; m < 4; ++m) {
                v16bf a = load_a_tile(sA + m * 16 * K1, K1, ks, lane);
                acc[m] = wmma_bf16(a, b_cur, acc[m]);
            }
            b_cur = b_next;
        }
#pragma unroll
        for (int m = 0; m < 4; ++m) {           // peeled last k-step
            v16bf a = load_a_tile(sA + m * 16 * K1, K1, K1 - 32, lane);
            acc[m] = wmma_bf16(a, b_cur, acc[m]);
        }
        const int n = ct * 16 + n_lo;
        const float bias = b1[n];
#pragma unroll
        for (int m = 0; m < 4; ++m) {
#pragma unroll
            for (int v = 0; v < 8; ++v) {
                float val = acc[m][v] + bias;
                val = val > 0.f ? val : 0.f;
                sH[(size_t)(m * 16 + v + 8 * h) * MLPSZ + n] = f32_to_bf16(val);
            }
        }
    }
}

__device__ __forceinline__ void mlp_layer2_block(
    const unsigned short* sH, const unsigned short* W2t /*[16][512]*/,
    const float* b2, float* out, int out_cols,
    long long row0, long long nrows, int wave, int lane)
{
    const int n_lo = lane & 15, h = lane >> 4;
    const unsigned short* sHw = sH + (size_t)wave * 16 * MLPSZ;
    v8f acc = {};
    v16bf b_cur = load_b_tile(W2t, MLPSZ, 0, 0, lane);
#pragma unroll 1
    for (int ks = 0; ks < MLPSZ - 32; ks += 32) {
        v16bf b_next = load_b_tile(W2t, MLPSZ, ks + 32, 0, lane);
        v16bf a = load_a_tile(sHw, MLPSZ, ks, lane);
        acc = wmma_bf16(a, b_cur, acc);
        b_cur = b_next;
    }
    {
        v16bf a = load_a_tile(sHw, MLPSZ, MLPSZ - 32, lane);
        acc = wmma_bf16(a, b_cur, acc);
    }
    if (n_lo < out_cols) {
        const float bias = b2[n_lo];
#pragma unroll
        for (int v = 0; v < 8; ++v) {
            long long grow = row0 + wave * 16 + v + 8 * h;
            if (grow < nrows)
                out[grow * out_cols + n_lo] = acc[v] + bias;
        }
    }
}

// ---------------------------------------------------------------------------
// Stage 2: bond MLP on gathered endpoint pairs (K1 = 512), 64 bonds/block
// ---------------------------------------------------------------------------
__global__ __launch_bounds__(128) void bond_kernel(
    const unsigned short* __restrict__ ha_bf,
    const int* __restrict__ bond_index,
    const unsigned short* __restrict__ W1t, const float* __restrict__ b1,
    const unsigned short* __restrict__ W2t, const float* __restrict__ b2,
    float* __restrict__ out, int n_bonds, int out_cols)
{
    __shared__ __align__(16) unsigned short sA[64 * 512];   // 64 KB
    __shared__ __align__(16) unsigned short sH[64 * 512];   // 64 KB
    const int row0 = blockIdx.x * 64;
    const int t = threadIdx.x;
    {   // gather: thread -> (row = t>>1, endpoint = t&1), 256 bf16 each
        int r = t >> 1, half = t & 1;
        int grow = row0 + r;
        if (grow >= n_bonds) grow = n_bonds - 1;
        int atom = bond_index[2 * grow + half];
        const uint4* src = (const uint4*)(ha_bf + (size_t)atom * HSIZE);
        uint4* dst = (uint4*)(sA + r * 512 + half * 256);
#pragma unroll
        for (int i = 0; i < 32; ++i) dst[i] = src[i];
    }
    __syncthreads();
    const int wave = t >> 5, lane = t & 31;
    mlp_layer1_block<512>(sA, sH, W1t, b1, wave, lane);
    __syncthreads();
    mlp_layer2_block(sH, W2t, b2, out, out_cols, row0, n_bonds, wave, lane);
}

// ---------------------------------------------------------------------------
// Stage 3: uni MLP directly on ha rows (K1 = 256), 64 atoms/block
// ---------------------------------------------------------------------------
__global__ __launch_bounds__(128) void uni_kernel(
    const unsigned short* __restrict__ ha_bf,
    const unsigned short* __restrict__ W1t, const float* __restrict__ b1,
    const unsigned short* __restrict__ W2t, const float* __restrict__ b2,
    float* __restrict__ out, int n_rows)
{
    __shared__ __align__(16) unsigned short sA[64 * 256];   // 32 KB
    __shared__ __align__(16) unsigned short sH[64 * 512];   // 64 KB
    const int row0 = blockIdx.x * 64;
    const int t = threadIdx.x;
    {
        int r = t >> 1, half = t & 1;
        int grow = row0 + r;
        if (grow >= n_rows) grow = n_rows - 1;
        const uint4* src = (const uint4*)(ha_bf + (size_t)grow * HSIZE + half * 128);
        uint4* dst = (uint4*)(sA + r * 256 + half * 128);
#pragma unroll
        for (int i = 0; i < 16; ++i) dst[i] = src[i];
    }
    __syncthreads();
    const int wave = t >> 5, lane = t & 31;
    mlp_layer1_block<256>(sA, sH, W1t, b1, wave, lane);
    __syncthreads();
    mlp_layer2_block(sH, W2t, b2, out, 1, row0, n_rows, wave, lane);
}

// ---------------------------------------------------------------------------
// Stage 4: done MLP on hm (fp32 -> bf16 on stage-in, K1 = 256), 64 mols/block
// ---------------------------------------------------------------------------
__global__ __launch_bounds__(128) void done_kernel(
    const float* __restrict__ hm,
    const unsigned short* __restrict__ W1t, const float* __restrict__ b1,
    const unsigned short* __restrict__ W2t, const float* __restrict__ b2,
    float* __restrict__ out, int n_mols)
{
    __shared__ __align__(16) unsigned short sA[64 * 256];   // 32 KB
    __shared__ __align__(16) unsigned short sH[64 * 512];   // 64 KB
    const int row0 = blockIdx.x * 64;
    const int t = threadIdx.x;
    {
        int r = t >> 1, half = t & 1;
        int grow = row0 + r;
        int gro  = grow < n_mols ? grow : n_mols - 1;
        const float4* src = (const float4*)(hm + (size_t)gro * HSIZE + half * 128);
        uint4* dst = (uint4*)(sA + r * 256 + half * 128);
#pragma unroll
        for (int i = 0; i < 16; ++i) {
            float4 f0 = src[2 * i], f1 = src[2 * i + 1];
            if (grow >= n_mols) { f0 = make_float4(0,0,0,0); f1 = f0; }
            uint4 pk;
            pk.x = pack2_bf16(f0.x, f0.y); pk.y = pack2_bf16(f0.z, f0.w);
            pk.z = pack2_bf16(f1.x, f1.y); pk.w = pack2_bf16(f1.z, f1.w);
            dst[i] = pk;
        }
    }
    __syncthreads();
    const int wave = t >> 5, lane = t & 31;
    mlp_layer1_block<256>(sA, sH, W1t, b1, wave, lane);
    __syncthreads();
    mlp_layer2_block(sH, W2t, b2, out, 1, row0, n_mols, wave, lane);
}

// ---------------------------------------------------------------------------
extern "C" void kernel_launch(void* const* d_in, const int* in_sizes, int n_in,
                              void* d_out, int out_size, void* d_ws, size_t ws_size,
                              hipStream_t stream)
{
    const float* c_atom  = (const float*)d_in[0];
    const float* ha_prev = (const float*)d_in[1];
    const float* W_vv    = (const float*)d_in[2];
    const float* W_vc    = (const float*)d_in[3];
    const float* bond_W1 = (const float*)d_in[4];
    const float* bond_b1 = (const float*)d_in[5];
    const float* bond_W2 = (const float*)d_in[6];
    const float* bond_b2 = (const float*)d_in[7];
    const float* uni_W1  = (const float*)d_in[8];
    const float* uni_b1  = (const float*)d_in[9];
    const float* uni_W2  = (const float*)d_in[10];
    const float* uni_b2  = (const float*)d_in[11];
    const float* done_W1 = (const float*)d_in[12];
    const float* done_b1 = (const float*)d_in[13];
    const float* done_W2 = (const float*)d_in[14];
    const float* done_b2 = (const float*)d_in[15];
    const int*   mol_id     = (const int*)d_in[16];
    const int*   bond_index = (const int*)d_in[17];

    const int n_atoms = in_sizes[16];
    const int n_bonds = in_sizes[17] / 2;
    const int bs_out  = in_sizes[7];                               // 5
    const int n_mols  = out_size - n_bonds * bs_out - n_atoms;     // 1000

    // workspace carve-up
    char* ws = (char*)d_ws;
    size_t off = 0;
    auto carve = [&](size_t bytes) {
        char* p = ws + off; off = (off + bytes + 255) & ~(size_t)255; return p;
    };
    unsigned short* ha_bf = (unsigned short*)carve((size_t)n_atoms * HSIZE * 2);
    float*          hm    = (float*)carve((size_t)n_mols * HSIZE * 4);
    unsigned short* wcomb = (unsigned short*)carve(256 * 512 * 2);
    unsigned short* bW1t  = (unsigned short*)carve(512 * 512 * 2);
    unsigned short* bW2t  = (unsigned short*)carve(16 * 512 * 2);
    unsigned short* uW1t  = (unsigned short*)carve(512 * 256 * 2);
    unsigned short* uW2t  = (unsigned short*)carve(16 * 512 * 2);
    unsigned short* dW1t  = (unsigned short*)carve(512 * 256 * 2);
    unsigned short* dW2t  = (unsigned short*)carve(16 * 512 * 2);

    // weight prep (bf16, K-contiguous for WMMA B tiles)
    build_wcomb_bf16<<<(256*512 + 255)/256, 256, 0, stream>>>(W_vv, W_vc, wcomb);
    transpose_pad_bf16<<<(512*512 + 255)/256, 256, 0, stream>>>(bond_W1, bW1t, 512, 512, 512);
    transpose_pad_bf16<<<(16*512  + 255)/256, 256, 0, stream>>>(bond_W2, bW2t, 512, bs_out, 16);
    transpose_pad_bf16<<<(512*256 + 255)/256, 256, 0, stream>>>(uni_W1,  uW1t, 256, 512, 512);
    transpose_pad_bf16<<<(16*512  + 255)/256, 256, 0, stream>>>(uni_W2,  uW2t, 512, 1, 16);
    transpose_pad_bf16<<<(512*256 + 255)/256, 256, 0, stream>>>(done_W1, dW1t, 256, 512, 512);
    transpose_pad_bf16<<<(16*512  + 255)/256, 256, 0, stream>>>(done_W2, dW2t, 512, 1, 16);

    zero_f32<<<(n_mols * HSIZE + 255)/256, 256, 0, stream>>>(hm, n_mols * HSIZE);

    const int ha_blocks = (n_atoms + 63) / 64;
    ha_kernel<<<ha_blocks, 128, 0, stream>>>(ha_prev, c_atom, wcomb, mol_id,
                                             ha_bf, hm, n_atoms);

    float* out_bond = (float*)d_out;
    float* out_uni  = out_bond + (size_t)n_bonds * bs_out;
    float* out_done = out_uni + n_atoms;

    bond_kernel<<<(n_bonds + 63)/64, 128, 0, stream>>>(ha_bf, bond_index,
        bW1t, bond_b1, bW2t, bond_b2, out_bond, n_bonds, bs_out);

    uni_kernel<<<(n_atoms + 63)/64, 128, 0, stream>>>(ha_bf,
        uW1t, uni_b1, uW2t, uni_b2, out_uni, n_atoms);

    done_kernel<<<(n_mols + 63)/64, 128, 0, stream>>>(hm,
        dW1t, done_b1, dW2t, done_b2, out_done, n_mols);
}
